// LinguisticEncoder_74543452389993
// MI455X (gfx1250) — compile-verified, matrix operand order checked by reading, not run on hardware
//
#include <hip/hip_runtime.h>

// ---------------- model constants ----------------
#define BB 32
#define TT 128
#define DD 384
#define HH 6
#define DHH 64
#define NWW 32
#define MAXOUT 640
#define D4 1536
#define D2 768
#define NEGV (-1e9f)

#define OUT_MAIN ((long)BB * MAXOUT * DD)   // 7,864,320
#define OUT_CUM  (OUT_MAIN)                 // +32
#define OUT_WDUR (OUT_MAIN + 32)            // +1024

typedef __attribute__((ext_vector_type(16))) __bf16 v16bf;
typedef __attribute__((ext_vector_type(8)))  float  v8f;
typedef __attribute__((ext_vector_type(4)))  unsigned u32x4;
typedef unsigned short u16;

// ---------------- helpers ----------------
__device__ __forceinline__ u16 f2bf(float f) {
  unsigned u = __float_as_uint(f);
  unsigned r = u + 0x7FFFu + ((u >> 16) & 1u);
  return (u16)(r >> 16);
}
__device__ __forceinline__ float sigmoidf_(float x) { return 1.0f / (1.0f + __expf(-x)); }

// ---------------- weight transpose + bf16 convert: dst[n*K+k] = src[k*N+n] ----------------
__global__ void k_transpose_bf(const float* __restrict__ src, u16* __restrict__ dst, int Kd, int N) {
  long i = (long)blockIdx.x * blockDim.x + threadIdx.x;
  if (i >= (long)Kd * N) return;
  int n = (int)(i % N);
  int k = (int)(i / N);
  dst[(long)n * Kd + k] = f2bf(src[(long)k * N + n]);
}

// ---------------- generalized batched WMMA GEMM ----------------
// C[z][m][n] = act( (alpha*sum_k A[z][m][k]*B[z][n][k]) * nscale[n] + bias[n] )
// per-z offsets: (z/zdiv)*s?0 + (z%zdiv)*s?1. B is [n][k] row-major with row stride ldbn.
// One wave computes a 32x32 tile: 4 accumulators, 4 v_wmma_f32_16x16x32_bf16 per K-step.
// Fragments loaded as 2x16B vectors (requires lda/ldbn multiples of 8, bases 16B aligned).
union Frag { v16bf v; u32x4 q[2]; };

__global__ void k_wmma_gemm(const u16* __restrict__ A, const u16* __restrict__ Bm,
                            const float* __restrict__ bias, const float* __restrict__ nscale,
                            float* __restrict__ C, u16* __restrict__ Cbf,
                            int Kd, int zdiv,
                            long sA0, long sA1, int lda,
                            long sB0, long sB1, int ldbn,
                            long sC0, long sC1, int ldc,
                            int act, float alpha) {
  int z = blockIdx.z;
  int zo = z / zdiv, zi = z - zo * zdiv;
  const u16* Ap = A + zo * sA0 + zi * sA1;
  const u16* Bp = Bm + zo * sB0 + zi * sB1;
  long cbase = zo * sC0 + zi * sC1;
  int n0 = blockIdx.x * 32;
  int m0 = blockIdx.y * 32;
  int lane = threadIdx.x & 31;
  int half = lane >> 4;
  int rc = lane & 15;

  v8f a00, a01, a10, a11;
#pragma unroll
  for (int i = 0; i < 8; i++) { a00[i] = 0.f; a01[i] = 0.f; a10[i] = 0.f; a11[i] = 0.f; }

  const u16* ar0 = Ap + (long)(m0 + rc) * lda + 8 * half;
  const u16* ar1 = Ap + (long)(m0 + 16 + rc) * lda + 8 * half;
  const u16* br0 = Bp + (long)(n0 + rc) * ldbn + 8 * half;
  const u16* br1 = Bp + (long)(n0 + 16 + rc) * ldbn + 8 * half;

  for (int k0 = 0; k0 < Kd; k0 += 32) {
    Frag fa0, fa1, fb0, fb1;
    const u32x4* p;
    p = (const u32x4*)(ar0 + k0); fa0.q[0] = p[0]; fa0.q[1] = p[2];
    p = (const u32x4*)(ar1 + k0); fa1.q[0] = p[0]; fa1.q[1] = p[2];
    p = (const u32x4*)(br0 + k0); fb0.q[0] = p[0]; fb0.q[1] = p[2];
    p = (const u32x4*)(br1 + k0); fb1.q[0] = p[0]; fb1.q[1] = p[2];
    a00 = __builtin_amdgcn_wmma_f32_16x16x32_bf16(false, fa0.v, false, fb0.v, (short)0, a00, false, false);
    a01 = __builtin_amdgcn_wmma_f32_16x16x32_bf16(false, fa0.v, false, fb1.v, (short)0, a01, false, false);
    a10 = __builtin_amdgcn_wmma_f32_16x16x32_bf16(false, fa1.v, false, fb0.v, (short)0, a10, false, false);
    a11 = __builtin_amdgcn_wmma_f32_16x16x32_bf16(false, fa1.v, false, fb1.v, (short)0, a11, false, false);
  }

#pragma unroll
  for (int tn = 0; tn < 2; tn++) {
    int col = n0 + tn * 16 + rc;
    float sc = nscale ? nscale[col] : 1.0f;
    float bs = bias ? bias[col] : 0.0f;
#pragma unroll
    for (int tm = 0; tm < 2; tm++) {
#pragma unroll
      for (int r = 0; r < 8; r++) {
        float v;
        if (tm == 0) v = (tn == 0) ? a00[r] : a01[r];
        else         v = (tn == 0) ? a10[r] : a11[r];
        v = alpha * v;
        v = v * sc + bs;
        if (act == 1)      v = v * sigmoidf_(v);   // silu
        else if (act == 2) v = fmaxf(v, 0.0f);     // relu
        int m = m0 + tm * 16 + r + 8 * half;
        long idx = cbase + (long)m * ldc + col;
        if (C)   C[idx] = v;
        if (Cbf) Cbf[idx] = f2bf(v);
      }
    }
  }
}

// ---------------- elementwise / reduction kernels ----------------
__global__ void k_embed(const int* __restrict__ tok, const float* __restrict__ emb,
                        float* __restrict__ x, u16* __restrict__ xbf) {
  long i = (long)blockIdx.x * blockDim.x + threadIdx.x;
  if (i >= (long)BB * TT * DD) return;
  int d = (int)(i % DD);
  long r = i / DD;
  float v = emb[(long)tok[r] * DD + d];
  x[i] = v; xbf[i] = f2bf(v);
}

// key-masked softmax; one wave per row; writes bf16 probs
__global__ void k_softmax(const float* __restrict__ S, u16* __restrict__ P,
                          const int* __restrict__ lens, int Tq, int Tk) {
  long row = blockIdx.x;
  int b = (int)(row / ((long)HH * Tq));
  int len = lens[b];
  const float* s = S + row * Tk;
  u16* p = P + row * Tk;
  int lane = threadIdx.x;
  float m = -3.4e38f;
  for (int j = lane; j < Tk; j += 32) {
    float v = (j < len) ? s[j] : NEGV;
    m = fmaxf(m, v);
  }
  for (int o = 16; o > 0; o >>= 1) m = fmaxf(m, __shfl_xor(m, o, 32));
  float sum = 0.0f;
  for (int j = lane; j < Tk; j += 32) {
    float v = (j < len) ? s[j] : NEGV;
    sum += __expf(v - m);
  }
  for (int o = 16; o > 0; o >>= 1) sum += __shfl_xor(sum, o, 32);
  float inv = 1.0f / sum;
  for (int j = lane; j < Tk; j += 32) {
    float v = (j < len) ? s[j] : NEGV;
    p[j] = f2bf(__expf(v - m) * inv);
  }
}

// x = LN(xin + delta); writes f32 + bf16. One block (128 thr) per row, D=384.
__global__ void k_ln(const float* __restrict__ xin, const float* __restrict__ delta,
                     const float* __restrict__ g, const float* __restrict__ bb,
                     float* __restrict__ xout, u16* __restrict__ xbf) {
  long row = blockIdx.x;
  int t = threadIdx.x;
  __shared__ float red[128];
  float loc[3];
  float s = 0.0f;
#pragma unroll
  for (int i = 0; i < 3; i++) {
    long idx = row * DD + t + i * 128;
    float v = xin[idx] + delta[idx];
    loc[i] = v; s += v;
  }
  red[t] = s; __syncthreads();
  for (int o = 64; o > 0; o >>= 1) { if (t < o) red[t] += red[t + o]; __syncthreads(); }
  float mean = red[0] / DD; __syncthreads();
  float vs = 0.0f;
#pragma unroll
  for (int i = 0; i < 3; i++) { float d2 = loc[i] - mean; vs += d2 * d2; }
  red[t] = vs; __syncthreads();
  for (int o = 64; o > 0; o >>= 1) { if (t < o) red[t] += red[t + o]; __syncthreads(); }
  float inv = rsqrtf(red[0] / DD + 1e-5f);
#pragma unroll
  for (int i = 0; i < 3; i++) {
    int c = t + i * 128;
    float v = (loc[i] - mean) * inv * g[c] + bb[c];
    xout[row * DD + c] = v;
    xbf[row * DD + c] = f2bf(v);
  }
}

// masked bf16 copy: zero padded time steps (conv-module input)
__global__ void k_maskbf(const float* __restrict__ x, u16* __restrict__ o,
                         const int* __restrict__ lens, int SEQ) {
  long i = (long)blockIdx.x * blockDim.x + threadIdx.x;
  if (i >= (long)BB * SEQ * DD) return;
  long r = i / DD;
  int t = (int)(r % SEQ);
  int b = (int)(r / SEQ);
  o[i] = (t < lens[b]) ? f2bf(x[i]) : (u16)0;
}

__global__ void k_glu(const float* __restrict__ h, float* __restrict__ o, long rows) {
  long i = (long)blockIdx.x * blockDim.x + threadIdx.x;
  if (i >= rows * DD) return;
  long r = i / DD; int d = (int)(i % DD);
  float a = h[r * D2 + d];
  float g = h[r * D2 + DD + d];
  o[i] = a * sigmoidf_(g);
}

// depthwise conv (K=9, zero pad 4) + silu -> bf16
__global__ void k_dwconv(const float* __restrict__ x, const float* __restrict__ w,
                         const float* __restrict__ bb, u16* __restrict__ o, int SEQ) {
  long i = (long)blockIdx.x * blockDim.x + threadIdx.x;
  if (i >= (long)BB * SEQ * DD) return;
  int d = (int)(i % DD);
  long r = i / DD;
  int t = (int)(r % SEQ);
  int b = (int)(r / SEQ);
  float s = bb[d];
#pragma unroll
  for (int k = 0; k < 9; k++) {
    int tt = t + k - 4;
    if (tt >= 0 && tt < SEQ) s += x[((long)b * SEQ + tt) * DD + d] * w[k * DD + d];
  }
  o[i] = f2bf(s * sigmoidf_(s));
}

__global__ void k_zero_u16(u16* __restrict__ p, long n) {
  long i = (long)blockIdx.x * blockDim.x + threadIdx.x;
  if (i < n) p[i] = 0;
}

// copy x[B,T,D] into interior of padded bf16 buffer [B, T+8, D]
__global__ void k_padfill(const float* __restrict__ x, u16* __restrict__ xp) {
  long i = (long)blockIdx.x * blockDim.x + threadIdx.x;
  if (i >= (long)BB * TT * DD) return;
  int d = (int)(i % DD);
  long r = i / DD;
  int t = (int)(r % TT);
  int b = (int)(r / TT);
  xp[((long)b * (TT + 8) + t + 4) * DD + d] = f2bf(x[i]);
}

// final dp conv (3,D,1), pad 1, masked to 0 at padded positions
__global__ void k_c3(const float* __restrict__ h, const float* __restrict__ w,
                     const float* __restrict__ cb, const int* __restrict__ lens,
                     float* __restrict__ pd) {
  int bt = blockIdx.x;
  int b = bt / TT, t = bt % TT;
  __shared__ float red[128];
  float s = 0.0f;
  for (int tap = 0; tap < 3; tap++) {
    int tt = t + tap - 1;
    if (tt < 0 || tt >= TT) continue;
    for (int d = threadIdx.x; d < DD; d += 128)
      s += h[((long)b * TT + tt) * DD + d] * w[tap * DD + d];
  }
  red[threadIdx.x] = s; __syncthreads();
  for (int o = 64; o > 0; o >>= 1) { if (threadIdx.x < o) red[threadIdx.x] += red[threadIdx.x + o]; __syncthreads(); }
  if (threadIdx.x == 0) {
    float v = red[0] + cb[0];
    pd[bt] = (t < lens[b]) ? v : 0.0f;
  }
}

__global__ void k_wlens(const int* __restrict__ widx, int* __restrict__ wl) {
  int b = blockIdx.x;
  int lane = threadIdx.x;
  __shared__ int red[128];
  int m = 0;
  for (int t = lane; t < TT; t += 128) m = max(m, widx[b * TT + t]);
  red[lane] = m; __syncthreads();
  for (int o = 64; o > 0; o >>= 1) { if (lane < o) red[lane] = max(red[lane], red[lane + o]); __syncthreads(); }
  if (lane == 0) wl[b] = red[0] + 1;
}

// segment pooling: mean of p_enc per word, sum of exp(p_durs) per word
__global__ void k_pool(const float* __restrict__ x, const float* __restrict__ pd,
                       const int* __restrict__ widx, float* __restrict__ wemb,
                       u16* __restrict__ wembbf, float* __restrict__ wdur_out) {
  int w = blockIdx.x, b = blockIdx.y;
  __shared__ float cnt_s;
  if (threadIdx.x == 0) {
    int c = 0; float wd = 0.0f;
    for (int t = 0; t < TT; t++)
      if (widx[b * TT + t] == w) { c++; wd += __expf(pd[b * TT + t]); }
    cnt_s = (float)c;
    wdur_out[b * NWW + w] = wd;
  }
  __syncthreads();
  float inv = 1.0f / fmaxf(cnt_s, 1.0f);
  for (int d = threadIdx.x; d < DD; d += blockDim.x) {
    float s = 0.0f;
    for (int t = 0; t < TT; t++)
      if (widx[b * TT + t] == w) s += x[((long)b * TT + t) * DD + d];
    s *= inv;
    long idx = ((long)b * NWW + w) * DD + d;
    wemb[idx] = s; wembbf[idx] = f2bf(s);
  }
}

__global__ void k_cumout(const int* __restrict__ wdur, const int* __restrict__ wl,
                         float* __restrict__ outc) {
  int b = threadIdx.x;
  if (b >= BB) return;
  int L = wl[b];
  int tot = 0;
  for (int w = 0; w < NWW; w++) if (w < L) tot += wdur[b * NWW + w];
  outc[b] = (float)tot;
}

// length regulator: expd[b,t] = (t < total) ? y[b, idx(t)] : 0 (bf16 for Q gemm)
__global__ void k_regulate(const float* __restrict__ y, const int* __restrict__ wdur,
                           const int* __restrict__ wl, u16* __restrict__ expd) {
  int t = blockIdx.x, b = blockIdx.y;
  __shared__ int idx_s, val_s;
  if (threadIdx.x == 0) {
    int L = wl[b];
    int c = 0, cnt = 0;
    for (int w = 0; w < NWW; w++) {
      c += (w < L) ? wdur[b * NWW + w] : 0;
      if (c <= t) cnt++;
    }
    idx_s = min(cnt, NWW - 1);
    val_s = (t < c) ? 1 : 0;
  }
  __syncthreads();
  int id = idx_s, val = val_s;
  for (int d = threadIdx.x; d < DD; d += blockDim.x) {
    float v = val ? y[((long)b * NWW + id) * DD + d] : 0.0f;
    expd[((long)b * MAXOUT + t) * DD + d] = f2bf(v);
  }
}

// v-transpose for attention: dst[(b*H+h)*DH*SEQ + dh*SEQ + t] = src[(b*SEQ+t)*rowstride + coloff + h*DH+dh]
__global__ void k_vt(const u16* __restrict__ src, u16* __restrict__ dst,
                     int SEQ, int rowstride, int coloff) {
  long i = (long)blockIdx.x * blockDim.x + threadIdx.x;
  if (i >= (long)BB * SEQ * DD) return;
  int d = (int)(i % DD);
  long r = i / DD;
  int t = (int)(r % SEQ);
  int b = (int)(r / SEQ);
  int h = d / DHH, dh = d % DHH;
  dst[(((long)b * HH + h) * DHH + dh) * SEQ + t] = src[r * rowstride + coloff + d];
}

// concat 2 or 3 bias vectors of length D
__global__ void k_cat3(const float* __restrict__ a, const float* __restrict__ b,
                       const float* __restrict__ c, float* __restrict__ o, int nparts) {
  int i = blockIdx.x * blockDim.x + threadIdx.x;
  if (i >= nparts * DD) return;
  int part = i / DD, d = i - part * DD;
  const float* s = (part == 0) ? a : ((part == 1) ? b : c);
  o[i] = s[d];
}

// effective BN bias: o[d] = cb[d]*g[d] + bnb[d]
__global__ void k_effb(const float* __restrict__ cb, const float* __restrict__ g,
                       const float* __restrict__ bnb, float* __restrict__ o) {
  int i = blockIdx.x * blockDim.x + threadIdx.x;
  if (i < DD) o[i] = cb[i] * g[i] + bnb[i];
}

// ---------------- host side ----------------
struct BW {
  u16 *wqkv, *wo, *f1w1, *f1w2, *f2w1, *f2w2, *pw1, *pw2;
  float* qkvb;
};

static void gemm(hipStream_t st, const u16* A, const u16* B, const float* bias,
                 const float* nscale, float* C, u16* Cbf,
                 int M, int N, int Kd, int Z, int zdiv,
                 long sA0, long sA1, int lda,
                 long sB0, long sB1, int ldbn,
                 long sC0, long sC1, int ldc, int act, float alpha) {
  dim3 g(N / 32, M / 32, Z), b(32, 1, 1);
  k_wmma_gemm<<<g, b, 0, st>>>(A, B, bias, nscale, C, Cbf, Kd, zdiv,
                               sA0, sA1, lda, sB0, sB1, ldbn, sC0, sC1, ldc, act, alpha);
}

struct Scratch {
  u16 *qkvbf, *vT, *obf, *h1bf, *dwbf, *mxbf, *Pbf;
  float *S, *delta, *pw1out, *hglu;
};

static void run_block(hipStream_t st, void* const* d_in, int pb, const BW& w,
                      Scratch& c, float* x, u16* xbf, int SEQ, const int* lens) {
  auto F = [&](int i) { return (const float*)d_in[i]; };
  int ROWS = BB * SEQ;
  int ZH = BB * HH;
  long e = (long)ROWS * DD;
  // --- MHA: fused QKV projection ---
  gemm(st, xbf, w.wqkv, w.qkvb, nullptr, nullptr, c.qkvbf, ROWS, 3 * DD, DD, 1, 1,
       0, 0, DD, 0, 0, DD, 0, 0, 3 * DD, 0, 1.f);
  k_vt<<<(int)((e + 255) / 256), 256, 0, st>>>(c.qkvbf, c.vT, SEQ, 3 * DD, 2 * DD);
  gemm(st, c.qkvbf, c.qkvbf + DD, nullptr, nullptr, c.S, nullptr, SEQ, SEQ, DHH, ZH, HH,
       (long)SEQ * 3 * DD, DHH, 3 * DD,
       (long)SEQ * 3 * DD, DHH, 3 * DD,
       (long)HH * SEQ * SEQ, (long)SEQ * SEQ, SEQ, 0, 0.125f);
  k_softmax<<<ZH * SEQ, 32, 0, st>>>(c.S, c.Pbf, lens, SEQ, SEQ);
  gemm(st, c.Pbf, c.vT, nullptr, nullptr, nullptr, c.obf, SEQ, DHH, SEQ, ZH, HH,
       (long)HH * SEQ * SEQ, (long)SEQ * SEQ, SEQ,
       (long)HH * DHH * SEQ, (long)DHH * SEQ, SEQ,
       (long)SEQ * DD, DHH, DD, 0, 1.f);
  gemm(st, c.obf, w.wo, F(pb + 1), nullptr, c.delta, nullptr, ROWS, DD, DD, 1, 1,
       0, 0, DD, 0, 0, DD, 0, 0, DD, 0, 1.f);
  k_ln<<<ROWS, 128, 0, st>>>(x, c.delta, F(pb + 18), F(pb + 19), x, xbf);
  // --- FFN1 ---
  gemm(st, xbf, w.f1w1, F(pb + 10), nullptr, nullptr, c.h1bf, ROWS, D4, DD, 1, 1,
       0, 0, DD, 0, 0, DD, 0, 0, D4, 1, 1.f);
  gemm(st, c.h1bf, w.f1w2, F(pb + 11), nullptr, c.delta, nullptr, ROWS, DD, D4, 1, 1,
       0, 0, D4, 0, 0, D4, 0, 0, DD, 0, 1.f);
  k_ln<<<ROWS, 128, 0, st>>>(x, c.delta, F(pb + 20), F(pb + 21), x, xbf);
  // --- Conv module ---
  k_maskbf<<<(int)((e + 255) / 256), 256, 0, st>>>(x, c.mxbf, lens, SEQ);
  gemm(st, c.mxbf, w.pw1, F(pb + 26), nullptr, c.pw1out, nullptr, ROWS, D2, DD, 1, 1,
       0, 0, DD, 0, 0, DD, 0, 0, D2, 0, 1.f);
  k_glu<<<(int)((e + 255) / 256), 256, 0, st>>>(c.pw1out, c.hglu, ROWS);
  k_dwconv<<<(int)((e + 255) / 256), 256, 0, st>>>(c.hglu, F(pb + 9), F(pb + 8), c.dwbf, SEQ);
  gemm(st, c.dwbf, w.pw2, F(pb + 28), nullptr, c.delta, nullptr, ROWS, DD, DD, 1, 1,
       0, 0, DD, 0, 0, DD, 0, 0, DD, 0, 1.f);
  k_ln<<<ROWS, 128, 0, st>>>(x, c.delta, F(pb + 22), F(pb + 23), x, xbf);
  // --- FFN2 ---
  gemm(st, xbf, w.f2w1, F(pb + 14), nullptr, nullptr, c.h1bf, ROWS, D4, DD, 1, 1,
       0, 0, DD, 0, 0, DD, 0, 0, D4, 1, 1.f);
  gemm(st, c.h1bf, w.f2w2, F(pb + 15), nullptr, c.delta, nullptr, ROWS, DD, D4, 1, 1,
       0, 0, D4, 0, 0, D4, 0, 0, DD, 0, 1.f);
  k_ln<<<ROWS, 128, 0, st>>>(x, c.delta, F(pb + 24), F(pb + 25), x, xbf);
}

extern "C" void kernel_launch(void* const* d_in, const int* in_sizes, int n_in,
                              void* d_out, int out_size, void* d_ws, size_t ws_size,
                              hipStream_t stream) {
  (void)in_sizes; (void)n_in; (void)out_size; (void)ws_size;
  const int* tok  = (const int*)d_in[0];
  const int* tlen = (const int*)d_in[1];
  const int* widx = (const int*)d_in[2];
  const int* wdur = (const int*)d_in[3];
  auto F = [&](int i) { return (const float*)d_in[i]; };
  float* out_f = (float*)d_out;

  // ---- bump allocator over workspace ----
  size_t off = 0;
  auto ab = [&](size_t bytes) -> void* {
    void* p = (char*)d_ws + off;
    off = (off + bytes + 255) & ~(size_t)255;
    return p;
  };
  auto au16 = [&](long n) { return (u16*)ab((size_t)n * sizeof(u16)); };
  auto af32 = [&](long n) { return (float*)ab((size_t)n * sizeof(float)); };

  auto convp = [&](const float* src, u16* dst, int Kd, int N) {
    long n = (long)Kd * N;
    k_transpose_bf<<<(int)((n + 255) / 256), 256, 0, stream>>>(src, dst, Kd, N);
  };

  // ---- weight conversion (bf16, transposed [N,K]) ----
  // pytree order: dp(4..13), emb(14), ph blocks @15,45, w blocks @75,105, xattn @135
  int bases[4] = {15, 45, 75, 105};
  BW bw[4];
  for (int i = 0; i < 4; i++) {
    int pb = bases[i];
    bw[i].wqkv = au16((long)3 * DD * DD);
    convp(F(pb + 6), bw[i].wqkv,                 DD, DD);  // wq rows [0,D)
    convp(F(pb + 4), bw[i].wqkv + (long)DD * DD, DD, DD);  // wk rows [D,2D)
    convp(F(pb + 7), bw[i].wqkv + (long)2 * DD * DD, DD, DD);  // wv rows [2D,3D)
    bw[i].qkvb = af32(3 * DD);
    k_cat3<<<(3 * DD + 255) / 256, 256, 0, stream>>>(F(pb + 2), F(pb + 0), F(pb + 3), bw[i].qkvb, 3);
    bw[i].wo = au16((long)DD * DD);   convp(F(pb + 5),  bw[i].wo, DD, DD);
    bw[i].f1w1 = au16((long)DD * D4); convp(F(pb + 12), bw[i].f1w1, DD, D4);
    bw[i].f1w2 = au16((long)DD * D4); convp(F(pb + 13), bw[i].f1w2, D4, DD);
    bw[i].f2w1 = au16((long)DD * D4); convp(F(pb + 16), bw[i].f2w1, DD, D4);
    bw[i].f2w2 = au16((long)DD * D4); convp(F(pb + 17), bw[i].f2w2, D4, DD);
    bw[i].pw1 = au16((long)DD * D2);  convp(F(pb + 27), bw[i].pw1, DD, D2);
    bw[i].pw2 = au16((long)DD * DD);  convp(F(pb + 29), bw[i].pw2, DD, DD);
  }
  // xattn: q separate; k|v fused
  u16* xwq = au16((long)DD * DD);      convp(F(141), xwq, DD, DD);
  u16* xwkv = au16((long)2 * DD * DD);
  convp(F(139), xwkv, DD, DD);                    // wk rows [0,D)
  convp(F(142), xwkv + (long)DD * DD, DD, DD);    // wv rows [D,2D)
  u16* xwo = au16((long)DD * DD);      convp(F(140), xwo, DD, DD);
  float* kvb = af32(2 * DD);
  k_cat3<<<(2 * DD + 255) / 256, 256, 0, stream>>>(F(135), F(138), nullptr, kvb, 2);
  // dp conv weights as full im2col-transposed matrices [D, 9*D]
  u16* c1t = au16((long)9 * DD * DD);  convp(F(9),  c1t, 9 * DD, DD);
  u16* c2t = au16((long)9 * DD * DD);  convp(F(11), c2t, 9 * DD, DD);
  float* eff1 = af32(DD);
  float* eff2 = af32(DD);
  k_effb<<<2, 256, 0, stream>>>(F(8),  F(5), F(4), eff1);
  k_effb<<<2, 256, 0, stream>>>(F(10), F(7), F(6), eff2);

  // ---- activations ----
  long PH = (long)BB * TT * DD;             // 1,572,864
  long WR = (long)BB * NWW * DD;
  long XR = (long)BB * MAXOUT * DD;         // 7,864,320
  long SMAX = (long)BB * HH * MAXOUT * TT;  // 15,728,640
  float* x_p  = af32(PH); u16* x_pbf = au16(PH);
  float* x_w  = af32(WR); u16* x_wbf = au16(WR);
  Scratch c;
  c.qkvbf = au16((long)BB * TT * 3 * DD);
  c.vT = au16(PH);
  c.obf = au16(XR);                 // shared: attention out (ph/word) and cross
  c.S = af32(SMAX); c.Pbf = au16(SMAX);
  c.delta = af32(PH);
  c.h1bf = au16((long)BB * TT * D4);
  c.pw1out = af32((long)BB * TT * D2);
  c.hglu = af32(PH);
  c.dwbf = au16(PH); c.mxbf = au16(PH);
  u16* xpad  = au16((long)BB * (TT + 8) * DD);
  u16* h1pad = au16((long)BB * (TT + 8) * DD);
  float* h2   = af32(PH);
  float* pdur = af32((long)BB * TT);
  int* wl = (int*)ab(BB * sizeof(int));
  u16* kvbf = au16((long)BB * TT * 2 * DD);
  u16* expdbf = au16(XR);
  u16* q2bf   = au16(XR);

  // ---- 1) embedding ----
  k_embed<<<(int)((PH + 255) / 256), 256, 0, stream>>>(tok, F(14), x_p, x_pbf);

  // ---- 2) phoneme conformer blocks ----
  run_block(stream, d_in, 15, bw[0], c, x_p, x_pbf, TT, tlen);
  run_block(stream, d_in, 45, bw[1], c, x_p, x_pbf, TT, tlen);

  // ---- 3) duration predictor: each K=9 conv is ONE WMMA GEMM with K'=9*D over padded input ----
  long padE = (long)BB * (TT + 8) * DD;
  k_zero_u16<<<(int)((padE + 255) / 256), 256, 0, stream>>>(xpad, padE);
  k_padfill<<<(int)((PH + 255) / 256), 256, 0, stream>>>(x_p, xpad);
  k_zero_u16<<<(int)((padE + 255) / 256), 256, 0, stream>>>(h1pad, padE);
  gemm(stream, xpad, c1t, eff1, F(5), nullptr, h1pad + 4 * DD, TT, DD, 9 * DD, BB, 1,
       (long)(TT + 8) * DD, 0, DD, 0, 0, 9 * DD,
       (long)(TT + 8) * DD, 0, DD, 2, 1.f);
  gemm(stream, h1pad, c2t, eff2, F(7), h2, nullptr, TT, DD, 9 * DD, BB, 1,
       (long)(TT + 8) * DD, 0, DD, 0, 0, 9 * DD,
       (long)TT * DD, 0, DD, 2, 1.f);
  k_c3<<<BB * TT, 128, 0, stream>>>(h2, F(13), F(12), tlen, pdur);

  // ---- 4) word pooling ----
  k_wlens<<<BB, 128, 0, stream>>>(widx, wl);
  k_pool<<<dim3(NWW, BB), 128, 0, stream>>>(x_p, pdur, widx, x_w, x_wbf, out_f + OUT_WDUR);

  // ---- 5) word conformer blocks ----
  run_block(stream, d_in, 75,  bw[2], c, x_w, x_wbf, NWW, wl);
  run_block(stream, d_in, 105, bw[3], c, x_w, x_wbf, NWW, wl);

  // ---- 6) length regulator + cum output ----
  k_cumout<<<1, 32, 0, stream>>>(wdur, wl, out_f + OUT_CUM);
  k_regulate<<<dim3(MAXOUT, BB), 128, 0, stream>>>(x_w, wdur, wl, expdbf);

  // ---- 7) cross attention (word frames -> phonemes), writes d_out main ----
  gemm(stream, expdbf, xwq, F(137), nullptr, nullptr, q2bf, BB * MAXOUT, DD, DD, 1, 1,
       0, 0, DD, 0, 0, DD, 0, 0, DD, 0, 1.f);
  gemm(stream, x_pbf, xwkv, kvb, nullptr, nullptr, kvbf, BB * TT, 2 * DD, DD, 1, 1,
       0, 0, DD, 0, 0, DD, 0, 0, 2 * DD, 0, 1.f);
  k_vt<<<(int)((PH + 255) / 256), 256, 0, stream>>>(kvbf, c.vT, TT, 2 * DD, DD);
  gemm(stream, q2bf, kvbf, nullptr, nullptr, c.S, nullptr, MAXOUT, TT, DHH, BB * HH, HH,
       (long)MAXOUT * DD, DHH, DD,
       (long)TT * 2 * DD, DHH, 2 * DD,
       (long)HH * MAXOUT * TT, (long)MAXOUT * TT, TT, 0, 0.125f);
  k_softmax<<<BB * HH * MAXOUT, 32, 0, stream>>>(c.S, c.Pbf, tlen, MAXOUT, TT);
  gemm(stream, c.Pbf, c.vT, nullptr, nullptr, nullptr, c.obf, MAXOUT, DHH, TT, BB * HH, HH,
       (long)HH * MAXOUT * TT, (long)MAXOUT * TT, TT,
       (long)HH * DHH * TT, (long)DHH * TT, TT,
       (long)MAXOUT * DD, DHH, DD, 0, 1.f);
  gemm(stream, c.obf, xwo, F(136), nullptr, out_f, nullptr, BB * MAXOUT, DD, DD, 1, 1,
       0, 0, DD, 0, 0, DD, 0, 0, DD, 0, 1.f);
}
